// PerfPredSqz_42399917146170
// MI455X (gfx1250) — compile-verified, matrix operand order checked by reading, not run on hardware
//
#include <hip/hip_runtime.h>
#include <hip/hip_bf16.h>

// ---------------------------------------------------------------------------
// Performer (FAVOR+) encoder for MI455X (gfx1250, wave32, WMMA).
// All matmuls run through one bf16 WMMA TN-GEMM (v_wmma_f32_16x16x32_bf16),
// f32 accumulate, fused bias/residual/row-scale/GELU/tanh epilogue.
// Tile staging uses GLOBAL_LOAD_ASYNC_TO_LDS_B128 (ASYNCcnt) with
// double-buffered LDS: one barrier per K-step, fill overlaps WMMA.
// ---------------------------------------------------------------------------

typedef __attribute__((ext_vector_type(16))) __bf16 bf16x16;
typedef __attribute__((ext_vector_type(8)))  float  f32x8;

#define BD     32
#define SEQ0   1024
#define DM     512
#define NHEADS 8
#define DHD    64
#define MFEAT  266     // int(64*log(64))
#define MPADF  288     // MFEAT padded to mult of 32
#define FFD    2048
#define DEPTH  8

static const int WS_[DEPTH + 1] = {1024, 724, 512, 362, 256, 181, 128, 91, 64};

// ----------------------------- device helpers ------------------------------

__device__ __forceinline__ __bf16 f2bf(float f) {
  union { float f; unsigned u; } a; a.f = f;
  unsigned u = a.u + (0x7FFFu + ((a.u >> 16) & 1u));   // round-to-nearest-even
  union { unsigned short s; __bf16 b; } o; o.s = (unsigned short)(u >> 16);
  return o.b;
}
__device__ __forceinline__ float bf2f(__bf16 b) {
  union { unsigned short s; __bf16 b; } a; a.b = b;
  union { unsigned u; float f; } o; o.u = ((unsigned)a.s) << 16;
  return o.f;
}
__device__ __forceinline__ float gelu_exact(float x) {
  return 0.5f * x * (1.0f + erff(x * 0.70710678118654752f));
}

// raw 32-bit LDS byte offset of a __shared__ object (addrspace(3) ptrtoint)
__device__ __forceinline__ unsigned lds_off32(const __bf16* p) {
  return (unsigned)(unsigned long long)(__attribute__((address_space(3))) const __bf16*)p;
}

// async global->LDS copy of 32 bytes (two B128 transfers), tracked by ASYNCcnt
__device__ __forceinline__ void async_copy32(unsigned lds_byte, const __bf16* g) {
  asm volatile(
      "global_load_async_to_lds_b128 %0, %2, off\n\t"
      "global_load_async_to_lds_b128 %1, %2, off offset:16"
      :: "v"(lds_byte), "v"(lds_byte + 16u), "v"(g)
      : "memory");
}
__device__ __forceinline__ void wait_async0() {
  asm volatile("s_wait_asynccnt 0" ::: "memory");
}

union FragBF { uint4 u[2]; bf16x16 v; };

// ----------------------------- WMMA TN GEMM --------------------------------
// C[M,N] = act( A[M,K] * Bt[N,K]^T + bias (+resid) (*rowscale) )
// A,Bt bf16 (K-contiguous, K % 32 == 0 via buffer padding); C f32 or bf16.
// Two-level batching: z -> (z0 = z / b1cnt, z1 = z % b1cnt) with per-level
// element strides for every pointer (handles per-(batch,head) sub-views).

__global__ __launch_bounds__(128) void k_gemm(
    const __bf16* __restrict__ A, const __bf16* __restrict__ Bt,
    void* __restrict__ Cv,
    const float* __restrict__ bias, int bias_mode,      // 0 none,1 per-col,2 per-row
    const float* __restrict__ resid, const float* __restrict__ rscale,
    int Mtot, int Ntot, int K, int lda, int ldb, int ldc,
    int act, int c_bf16, int b1cnt,                      // act: 0 none,1 gelu,2 tanh
    long long aB0, long long aB1, long long bB0, long long bB1,
    long long cB0, long long cB1, long long rB0, long long rB1,
    long long sB0, long long sB1)
{
  const int z  = blockIdx.z;
  const int z0 = z / b1cnt, z1 = z - z0 * b1cnt;
  A  += z0 * aB0 + z1 * aB1;
  Bt += z0 * bB0 + z1 * bB1;
  const long long coff = z0 * cB0 + z1 * cB1;
  if (resid)  resid  += z0 * rB0 + z1 * rB1;
  if (rscale) rscale += z0 * sB0 + z1 * sB1;

  const int tileM = blockIdx.y * 64;
  const int tileN = blockIdx.x * 64;

  // double-buffered tiles; 80B row stride keeps 16B alignment, staggers banks
  __shared__ __bf16 As[2][64][40];
  __shared__ __bf16 Bs[2][64][40];

  const int tid  = threadIdx.x;
  const int wave = tid >> 5;
  const int lane = tid & 31;
  const int wm   = (wave >> 1) * 32;   // wave's M offset inside tile
  const int wn   = (wave & 1) * 32;    // wave's N offset inside tile
  const int lrow = tid >> 1;           // tile loader: row
  const int lk   = (tid & 1) * 16;     // tile loader: 16-half chunk
  const int frow = lane & 15;          // WMMA fragment row
  const int fkl  = (lane >> 4) * 8;    // WMMA fragment K-half select

  f32x8 acc[2][2] = {};

  // stage one 64x32 A-tile + 64x32 B-tile into LDS buffer `buf` (async)
  auto stage = [&](int buf, int k0) {
    const uint4 zz = {0, 0, 0, 0};
    const int gr = tileM + lrow;
    if (gr < Mtot) {
      async_copy32(lds_off32(&As[buf][lrow][lk]), A + (long long)gr * lda + k0 + lk);
    } else {
      *(uint4*)(&As[buf][lrow][lk])     = zz;
      *(uint4*)(&As[buf][lrow][lk + 8]) = zz;
    }
    const int gn = tileN + lrow;
    if (gn < Ntot) {
      async_copy32(lds_off32(&Bs[buf][lrow][lk]), Bt + (long long)gn * ldb + k0 + lk);
    } else {
      *(uint4*)(&Bs[buf][lrow][lk])     = zz;
      *(uint4*)(&Bs[buf][lrow][lk + 8]) = zz;
    }
  };

  stage(0, 0);
  int buf = 0;
  for (int k0 = 0; k0 < K; k0 += 32) {
    wait_async0();        // this wave's async fills done (tile k0 in LDS)
    __syncthreads();      // whole block: tile resident; prior-buffer reads retired
    if (k0 + 32 < K) stage(buf ^ 1, k0 + 32);   // overlap next fill with WMMAs

    FragBF af0, af1, bf0, bf1;
    {
      const __bf16* p0 = &As[buf][wm +      frow][fkl];
      af0.u[0] = *(const uint4*)p0;  af0.u[1] = *(const uint4*)(p0 + 16);
      const __bf16* p1 = &As[buf][wm + 16 + frow][fkl];
      af1.u[0] = *(const uint4*)p1;  af1.u[1] = *(const uint4*)(p1 + 16);
      const __bf16* q0 = &Bs[buf][wn +      frow][fkl];
      bf0.u[0] = *(const uint4*)q0;  bf0.u[1] = *(const uint4*)(q0 + 16);
      const __bf16* q1 = &Bs[buf][wn + 16 + frow][fkl];
      bf1.u[0] = *(const uint4*)q1;  bf1.u[1] = *(const uint4*)(q1 + 16);
    }
    acc[0][0] = __builtin_amdgcn_wmma_f32_16x16x32_bf16(false, af0.v, false, bf0.v, (short)0, acc[0][0], false, false);
    acc[0][1] = __builtin_amdgcn_wmma_f32_16x16x32_bf16(false, af0.v, false, bf1.v, (short)0, acc[0][1], false, false);
    acc[1][0] = __builtin_amdgcn_wmma_f32_16x16x32_bf16(false, af1.v, false, bf0.v, (short)0, acc[1][0], false, false);
    acc[1][1] = __builtin_amdgcn_wmma_f32_16x16x32_bf16(false, af1.v, false, bf1.v, (short)0, acc[1][1], false, false);
    buf ^= 1;
  }

  float*  C  = (float*)Cv  + coff;
  __bf16* Cb = (__bf16*)Cv + coff;
  for (int im = 0; im < 2; ++im) {
    for (int jn = 0; jn < 2; ++jn) {
      const int col = tileN + wn + jn * 16 + (lane & 15);
      if (col >= Ntot) continue;
      const int rbase = tileM + wm + im * 16 + (lane >> 4) * 8;
      f32x8 a = acc[im][jn];
#pragma unroll
      for (int r = 0; r < 8; ++r) {
        const int row = rbase + r;
        if (row < Mtot) {
          float v = a[r];
          if      (bias_mode == 1) v += bias[col];
          else if (bias_mode == 2) v += bias[row];
          if (rscale) v *= rscale[row];
          if (resid)  v += resid[(long long)row * ldc + col];
          if      (act == 1) v = gelu_exact(v);
          else if (act == 2) v = tanhf(v);
          const long long idx = (long long)row * ldc + col;
          if (c_bf16) Cb[idx] = f2bf(v);
          else        C[idx]  = v;
        }
      }
    }
  }
}

// ------------------------- elementwise / reductions ------------------------

__global__ void k_convert(const float* __restrict__ s, __bf16* __restrict__ d,
                          long long n, float sc) {
  long long i = (long long)blockIdx.x * blockDim.x + threadIdx.x;
  const long long st = (long long)gridDim.x * blockDim.x;
  for (; i < n; i += st) d[i] = f2bf(s[i] * sc);
}

// 2D convert with zero-padded tail columns (ld_src == Cpad)
__global__ void k_convert_pad(const float* __restrict__ s, __bf16* __restrict__ d,
                              long long n, int C, int Cpad) {
  long long i = (long long)blockIdx.x * blockDim.x + threadIdx.x;
  const long long st = (long long)gridDim.x * blockDim.x;
  for (; i < n; i += st) {
    const int c = (int)(i % Cpad);
    d[i] = (c < C) ? f2bf(s[i]) : f2bf(0.f);
  }
}

// Batched transpose + fp32->bf16: src slice (R x Cc, ld_src) -> dst (Cc x Rpad),
// rows [R, Rpad) zero-filled.  z -> (z0=z/b1, z1=z%b1) offsets the source.
__global__ void k_transpose(const float* __restrict__ src, __bf16* __restrict__ dst,
                            int R, int Cc, int lds, int Rpad, float scale, int b1,
                            long long sB0, long long sB1, long long dB, int tot) {
  const int e = blockIdx.x * 256 + threadIdx.x;
  if (e >= tot) return;
  const int z = blockIdx.z, z0 = z / b1, z1 = z - z0 * b1;
  const int c = e / Rpad, r = e - c * Rpad;
  float v = 0.f;
  if (r < R) v = src[z0 * sB0 + z1 * sB1 + (long long)r * lds + c] * scale;
  dst[(long long)z * dB + (long long)c * Rpad + r] = f2bf(v);
}

__global__ void k_posemb(float* __restrict__ P, int n) {
  const int e = blockIdx.x * 256 + threadIdx.x;
  if (e >= n * DM) return;
  const int t = e / DM, d = e - t * DM;
  const int j = (d < DM / 2) ? d : d - DM / 2;
  const float invf = powf(10000.f, -(2.f * j) / (float)DM);
  const float ang = (float)t * invf;
  P[e] = (d < DM / 2) ? sinf(ang) : cosf(ang);
}

// rotary (layer 0): per pair within each 64-wide head slice
__global__ void k_rotary(float* __restrict__ X, long long npairs, int n) {
  long long p = (long long)blockIdx.x * blockDim.x + threadIdx.x;
  const long long st = (long long)gridDim.x * blockDim.x;
  for (; p < npairs; p += st) {
    const long long idx = p * 2;
    const long long row = idx / DM;
    const int t = (int)(row % n);
    const int j = ((int)(idx & 63)) >> 1;
    const float invf = powf(10000.f, -(2.f * j) / (float)DHD);
    const float ang = (float)t * invf;
    const float s = sinf(ang), c = cosf(ang);
    const float x0 = X[idx], x1 = X[idx + 1];
    X[idx]     = x0 * c - x1 * s;
    X[idx + 1] = x1 * c + x0 * s;
  }
}

__global__ __launch_bounds__(256) void k_layernorm(const float* __restrict__ X,
    __bf16* __restrict__ Y, const float* __restrict__ sc, const float* __restrict__ bi) {
  const long long row = blockIdx.x;
  const float* x = X + row * DM;
  const int tid = threadIdx.x;
  const float v0 = x[tid], v1 = x[tid + 256];
  __shared__ float red[256];
  red[tid] = v0 + v1; __syncthreads();
  for (int s = 128; s > 0; s >>= 1) { if (tid < s) red[tid] += red[tid + s]; __syncthreads(); }
  const float mu = red[0] * (1.f / (float)DM);
  __syncthreads();
  const float d0 = v0 - mu, d1 = v1 - mu;
  red[tid] = d0 * d0 + d1 * d1; __syncthreads();
  for (int s = 128; s > 0; s >>= 1) { if (tid < s) red[tid] += red[tid + s]; __syncthreads(); }
  const float rstd = rsqrtf(red[0] * (1.f / (float)DM) + 1e-5f);
  __bf16* y = Y + row * DM;
  y[tid]       = f2bf(d0 * rstd * sc[tid] + bi[tid]);
  y[tid + 256] = f2bf(d1 * rstd * sc[tid + 256] + bi[tid + 256]);
}

// diag[bh*n + t] = 0.5*dn^2 * sum_d q[b,t,h*64+d]^2
__global__ __launch_bounds__(64) void k_diag(const float* __restrict__ Q,
                                             float* __restrict__ Dg, int n, float hdn2) {
  const int rowg = blockIdx.x;       // b*n + t
  const int h = blockIdx.y;
  const float v = Q[(long long)rowg * DM + h * DHD + threadIdx.x];
  __shared__ float red[64];
  red[threadIdx.x] = v * v; __syncthreads();
  for (int s = 32; s > 0; s >>= 1) { if (threadIdx.x < s) red[threadIdx.x] += red[threadIdx.x + s]; __syncthreads(); }
  if (threadIdx.x == 0) {
    const int b = rowg / n, t = rowg - b * n;
    Dg[((long long)(b * NHEADS + h)) * n + t] = hdn2 * red[0];
  }
}

// qp = ratio*(exp(dd - diag - rowmax) + eps), zero-padded to MPADF, bf16 out
__global__ __launch_bounds__(256) void k_qp(const float* __restrict__ dd,
    const float* __restrict__ diag, __bf16* __restrict__ qp, float ratio) {
  const long long row = blockIdx.x;      // bh*n + t
  const float* d = dd + row * MPADF;
  const int tid = threadIdx.x;
  const float a0 = d[tid];                                       // tid < 256 < MFEAT
  const float a1 = (tid + 256 < MFEAT) ? d[tid + 256] : -3.4e38f;
  __shared__ float red[256];
  red[tid] = fmaxf(a0, a1); __syncthreads();
  for (int s = 128; s > 0; s >>= 1) { if (tid < s) red[tid] = fmaxf(red[tid], red[tid + s]); __syncthreads(); }
  const float mx = red[0];
  const float dg = diag[row];
  __bf16* o = qp + row * MPADF;
  o[tid] = f2bf(ratio * (expf(a0 - dg - mx) + 1e-4f));
  const int c2 = tid + 256;
  if (c2 < MPADF) {
    const float v = (c2 < MFEAT) ? ratio * (expf(a1 - dg - mx) + 1e-4f) : 0.f;
    o[c2] = f2bf(v);
  }
}

__global__ __launch_bounds__(256) void k_gmax1(const float* __restrict__ dd,
    float* __restrict__ part, long long total, int n, int npad) {
  long long i = (long long)blockIdx.x * 256 + threadIdx.x;
  const long long st = (long long)gridDim.x * 256;
  float m = -3.4e38f;
  for (; i < total; i += st) {
    const long long rm = i / n;          // bh*MFEAT + mf
    const int t = (int)(i - rm * n);
    m = fmaxf(m, dd[rm * npad + t]);
  }
  __shared__ float red[256];
  red[threadIdx.x] = m; __syncthreads();
  for (int s = 128; s > 0; s >>= 1) { if (threadIdx.x < s) red[threadIdx.x] = fmaxf(red[threadIdx.x], red[threadIdx.x + s]); __syncthreads(); }
  if (threadIdx.x == 0) part[blockIdx.x] = red[0];
}

__global__ __launch_bounds__(256) void k_gmax2(const float* __restrict__ part,
                                               float* __restrict__ gm, int cnt) {
  float m = -3.4e38f;
  for (int i = threadIdx.x; i < cnt; i += 256) m = fmaxf(m, part[i]);
  __shared__ float red[256];
  red[threadIdx.x] = m; __syncthreads();
  for (int s = 128; s > 0; s >>= 1) { if (threadIdx.x < s) red[threadIdx.x] = fmaxf(red[threadIdx.x], red[threadIdx.x + s]); __syncthreads(); }
  if (threadIdx.x == 0) gm[0] = red[0];
}

// kp (stored transposed: bh x MFEAT x npad) with zero pad cols, bf16 out
__global__ void k_kp(const float* __restrict__ dd, const float* __restrict__ diagk,
                     const float* __restrict__ gm, __bf16* __restrict__ kp,
                     long long total, int n, int npad, float ratio) {
  long long i = (long long)blockIdx.x * blockDim.x + threadIdx.x;
  const long long st = (long long)gridDim.x * blockDim.x;
  const float g = gm[0];
  for (; i < total; i += st) {
    const long long rm = i / npad;        // bh*MFEAT + mf
    const int t = (int)(i - rm * npad);
    float v = 0.f;
    if (t < n) {
      const long long bh = rm / MFEAT;
      v = ratio * (expf(dd[i] - diagk[bh * n + t] - g) + 1e-4f);
    }
    kp[i] = f2bf(v);
  }
}

__global__ __launch_bounds__(256) void k_rowsum(const __bf16* __restrict__ kp,
                                                float* __restrict__ ks, int n, int npad) {
  const long long row = blockIdx.x;        // bh*MFEAT + mf
  const __bf16* p = kp + row * npad;
  float s = 0.f;
  for (int t = threadIdx.x; t < n; t += 256) s += bf2f(p[t]);
  __shared__ float red[256];
  red[threadIdx.x] = s; __syncthreads();
  for (int st = 128; st > 0; st >>= 1) { if (threadIdx.x < st) red[threadIdx.x] += red[threadIdx.x + st]; __syncthreads(); }
  if (threadIdx.x == 0) ks[row] = red[0];
}

__global__ __launch_bounds__(256) void k_dinv(const __bf16* __restrict__ qp,
    const float* __restrict__ ks, float* __restrict__ dinv, int n) {
  const long long row = blockIdx.x;        // bh*n + t
  const long long bh = row / n;
  const __bf16* p = qp + row * MPADF;
  const float* k = ks + bh * MFEAT;
  float s = 0.f;
  for (int m = threadIdx.x; m < MFEAT; m += 256) s += bf2f(p[m]) * k[m];
  __shared__ float red[256];
  red[threadIdx.x] = s; __syncthreads();
  for (int st = 128; st > 0; st >>= 1) { if (threadIdx.x < st) red[threadIdx.x] += red[threadIdx.x + st]; __syncthreads(); }
  if (threadIdx.x == 0) dinv[row] = 1.f / red[0];
}

// ------------------------------- host side ---------------------------------

extern "C" void kernel_launch(void* const* d_in, const int* in_sizes, int n_in,
                              void* d_out, int out_size, void* d_ws, size_t ws_size,
                              hipStream_t stream) {
  (void)in_sizes; (void)n_in; (void)out_size; (void)ws_size;

  const float* x_in  = (const float*)d_in[0];
  const float* tokw  = (const float*)d_in[1];
  const float* tokb  = (const float*)d_in[2];
  const float* ln1s  = (const float*)d_in[3];
  const float* ln1b  = (const float*)d_in[4];
  const float* wq    = (const float*)d_in[5];
  const float* bq    = (const float*)d_in[6];
  const float* wk    = (const float*)d_in[7];
  const float* bk    = (const float*)d_in[8];
  const float* wv    = (const float*)d_in[9];
  const float* bv    = (const float*)d_in[10];
  const float* wo    = (const float*)d_in[11];
  const float* bo    = (const float*)d_in[12];
  const float* proj  = (const float*)d_in[13];
  const float* ln2s  = (const float*)d_in[14];
  const float* ln2b  = (const float*)d_in[15];
  const float* ffw1  = (const float*)d_in[16];
  const float* ffb1  = (const float*)d_in[17];
  const float* ffw2  = (const float*)d_in[18];
  const float* ffb2  = (const float*)d_in[19];

  // deterministic bump allocator over d_ws
  char* wsp = (char*)d_ws;
  size_t off = 0;
  auto alloc = [&](size_t bytes) -> void* {
    off = (off + 255) & ~(size_t)255;
    void* p = wsp + off;
    off += bytes;
    return p;
  };

  const long long ND  = (long long)BD * SEQ0 * DM;        // max activation elems
  float*  f_x1   = (float*) alloc(4 * ND);
  float*  f_x2   = (float*) alloc(4 * ND);
  float*  f_q    = (float*) alloc(4 * ND);
  float*  f_k    = (float*) alloc(4 * ND);
  float*  f_v    = (float*) alloc(4 * ND);
  float*  f_attn = (float*) alloc(4 * ND);
  __bf16* bf_h   = (__bf16*)alloc(2 * ND);
  __bf16* bf_opA = (__bf16*)alloc(2 * ND);
  float*  f_dd   = (float*) alloc(4ll * BD * NHEADS * SEQ0 * MPADF);
  __bf16* bf_qp  = (__bf16*)alloc(2ll * BD * NHEADS * SEQ0 * MPADF);
  __bf16* bf_kp  = (__bf16*)alloc(2ll * BD * NHEADS * MFEAT * SEQ0);
  __bf16* bf_vT  = (__bf16*)alloc(2ll * BD * NHEADS * DHD * SEQ0);
  __bf16* bf_xT  = (__bf16*)alloc(2ll * BD * DM * SEQ0);
  float*  f_ctx  = (float*) alloc(4ll * BD * NHEADS * DHD * MPADF);
  __bf16* bf_ctx = (__bf16*)alloc(2ll * BD * NHEADS * DHD * MPADF);
  float*  f_dgq  = (float*) alloc(4ll * BD * NHEADS * SEQ0);
  float*  f_dgk  = (float*) alloc(4ll * BD * NHEADS * SEQ0);
  float*  f_dinv = (float*) alloc(4ll * BD * NHEADS * SEQ0);
  float*  f_ksum = (float*) alloc(4ll * BD * NHEADS * MFEAT);
  float*  f_gpt  = (float*) alloc(4ll * 1024);
  float*  f_gm   = (float*) alloc(4ll * 4);
  float*  f_pos  = (float*) alloc(4ll * SEQ0 * DM);
  __bf16* bf_ff  = (__bf16*)alloc(2ll * BD * SEQ0 * FFD);
  __bf16* bf_wq  = (__bf16*)alloc(2ll * DM * DM);
  __bf16* bf_wk  = (__bf16*)alloc(2ll * DM * DM);
  __bf16* bf_wv  = (__bf16*)alloc(2ll * DM * DM);
  __bf16* bf_wo  = (__bf16*)alloc(2ll * DM * DM);
  __bf16* bf_f1  = (__bf16*)alloc(2ll * FFD * DM);
  __bf16* bf_f2  = (__bf16*)alloc(2ll * DM * FFD);
  __bf16* bf_lin = (__bf16*)alloc(2ll * 1024 * 1024);
  __bf16* bf_prj = (__bf16*)alloc(2ll * MFEAT * DHD);
  __bf16* bf_tok = (__bf16*)alloc(2ll * DM * DM);

  auto gemm = [&](const __bf16* A, const __bf16* Bt, void* C, const float* bias,
                  int bmode, const float* resid, const float* rscale,
                  int M, int N, int K, int lda, int ldb, int ldc,
                  int act, int cbf, int batches, int b1,
                  long long aB0, long long aB1, long long bB0, long long bB1,
                  long long cB0, long long cB1, long long rB0, long long rB1,
                  long long sB0, long long sB1) {
    dim3 g((N + 63) / 64, (M + 63) / 64, batches);
    k_gemm<<<g, dim3(128), 0, stream>>>(A, Bt, C, bias, bmode, resid, rscale,
        M, N, K, lda, ldb, ldc, act, cbf, b1,
        aB0, aB1, bB0, bB1, cB0, cB1, rB0, rB1, sB0, sB1);
  };
  auto conv = [&](const float* s, __bf16* d, long long cnt, float sc) {
    long long b = (cnt + 255) / 256; if (b > 8192) b = 8192;
    k_convert<<<(int)b, 256, 0, stream>>>(s, d, cnt, sc);
  };
  auto transp = [&](const float* s, __bf16* d, int R, int Cc, int lds, int Rpad,
                    float sc, int batches, int b1, long long sB0, long long sB1,
                    long long dB) {
    const int tot = Cc * Rpad;
    dim3 g((tot + 255) / 256, 1, batches);
    k_transpose<<<g, 256, 0, stream>>>(s, d, R, Cc, lds, Rpad, sc, b1, sB0, sB1, dB, tot);
  };

  const float dnv   = 0.35355339059327373f;   // 64^-0.25
  const float hdn2  = 0.0625f;                // 0.5 * dn^2
  const float ratio = 0.06131393394849658f;   // 266^-0.5

  // x = x @ tok_w + tok_b + pos
  k_posemb<<<(SEQ0 * DM + 255) / 256, 256, 0, stream>>>(f_pos, SEQ0);
  conv(x_in, bf_opA, ND, 1.f);
  transp(tokw, bf_tok, DM, DM, DM, DM, 1.f, 1, 1, 0, 0, 0);
  gemm(bf_opA, bf_tok, f_x1, tokb, 1, f_pos, nullptr,
       SEQ0, DM, DM, DM, DM, DM, 0, 0, BD, 1,
       (long long)SEQ0 * DM, 0, 0, 0, (long long)SEQ0 * DM, 0, 0, 0, 0, 0);

  for (int i = 0; i < DEPTH; ++i) {
    const int n    = WS_[i];
    const int npad = (n + 31) & ~31;
    const int nout = WS_[i + 1];
    const long long BN = (long long)BD * n;
    const long long BH = (long long)BD * NHEADS;

    // ---- LN1 -> bf16
    k_layernorm<<<(int)BN, 256, 0, stream>>>(f_x1, bf_h, ln1s + i * DM, ln1b + i * DM);

    // ---- per-layer weight staging (fp32 -> bf16, pre-transposed to N x K)
    transp(wq + (long long)i * DM * DM, bf_wq, DM, DM, DM, DM, 1.f, 1, 1, 0, 0, 0);
    transp(wk + (long long)i * DM * DM, bf_wk, DM, DM, DM, DM, 1.f, 1, 1, 0, 0, 0);
    transp(wv + (long long)i * DM * DM, bf_wv, DM, DM, DM, DM, 1.f, 1, 1, 0, 0, 0);
    transp(wo + (long long)i * DM * DM, bf_wo, DM, DM, DM, DM, 1.f, 1, 1, 0, 0, 0);
    conv(proj + (long long)i * MFEAT * DHD, bf_prj, (long long)MFEAT * DHD, 1.f);

    // ---- Q, K, V projections
    gemm(bf_h, bf_wq, f_q, bq + i * DM, 1, nullptr, nullptr,
         (int)BN, DM, DM, DM, DM, DM, 0, 0, 1, 1, 0,0,0,0,0,0,0,0,0,0);
    gemm(bf_h, bf_wk, f_k, bk + i * DM, 1, nullptr, nullptr,
         (int)BN, DM, DM, DM, DM, DM, 0, 0, 1, 1, 0,0,0,0,0,0,0,0,0,0);
    gemm(bf_h, bf_wv, f_v, bv + i * DM, 1, nullptr, nullptr,
         (int)BN, DM, DM, DM, DM, DM, 0, 0, 1, 1, 0,0,0,0,0,0,0,0,0,0);

    // ---- rotary (layer 0 only)
    if (i == 0) {
      const long long np = BN * DM / 2;
      k_rotary<<<8192, 256, 0, stream>>>(f_q, np, n);
      k_rotary<<<8192, 256, 0, stream>>>(f_k, np, n);
    }

    // ---- diag terms
    k_diag<<<dim3((int)BN, NHEADS), 64, 0, stream>>>(f_q, f_dgq, n, hdn2);
    k_diag<<<dim3((int)BN, NHEADS), 64, 0, stream>>>(f_k, f_dgk, n, hdn2);

    // ---- qp = softmax_kernel(q): dd = (q*dn) @ proj^T  per (b,h)
    conv(f_q, bf_opA, BN * DM, dnv);
    gemm(bf_opA, bf_prj, f_dd, nullptr, 0, nullptr, nullptr,
         n, MFEAT, DHD, DM, DHD, MPADF, 0, 0, (int)BH, NHEADS,
         (long long)n * DM, DHD, 0, 0,
         (long long)NHEADS * n * MPADF, (long long)n * MPADF, 0, 0, 0, 0);
    k_qp<<<(int)(BH * n), 256, 0, stream>>>(f_dd, f_dgq, bf_qp, ratio);

    // ---- kp (transposed): dd_k = proj @ (k*dn)^T, global max, exp
    conv(f_k, bf_opA, BN * DM, dnv);
    gemm(bf_prj, bf_opA, f_dd, nullptr, 0, nullptr, nullptr,
         MFEAT, n, DHD, DHD, DM, npad, 0, 0, (int)BH, NHEADS,
         0, 0, (long long)n * DM, DHD,
         (long long)NHEADS * MFEAT * npad, (long long)MFEAT * npad, 0, 0, 0, 0);
    k_gmax1<<<1024, 256, 0, stream>>>(f_dd, f_gpt, BH * MFEAT * n, n, npad);
    k_gmax2<<<1, 256, 0, stream>>>(f_gpt, f_gm, 1024);
    k_kp<<<4096, 256, 0, stream>>>(f_dd, f_dgk, f_gm, bf_kp,
                                   BH * MFEAT * npad, n, npad, ratio);
    k_rowsum<<<(int)(BH * MFEAT), 256, 0, stream>>>(bf_kp, f_ksum, n, npad);
    k_dinv<<<(int)(BH * n), 256, 0, stream>>>(bf_qp, f_ksum, f_dinv, n);

    // ---- ctx^T = v^T @ kp^T^T   (per b,h: DH x MFEAT, K = n)
    transp(f_v, bf_vT, n, DHD, DM, npad, 1.f, (int)BH, NHEADS,
           (long long)n * DM, DHD, (long long)DHD * npad);
    gemm(bf_vT, bf_kp, f_ctx, nullptr, 0, nullptr, nullptr,
         DHD, MFEAT, npad, npad, npad, MPADF, 0, 0, (int)BH, 1,
         (long long)DHD * npad, 0, (long long)MFEAT * npad, 0,
         (long long)DHD * MPADF, 0, 0, 0, 0, 0);
    {
      long long tot = BH * DHD * MPADF;
      k_convert_pad<<<4096, 256, 0, stream>>>(f_ctx, bf_ctx, tot, MFEAT, MPADF);
    }

    // ---- out = (qp @ ctx) * d_inv, heads written into (B,n,D) at col h*DH
    gemm(bf_qp, bf_ctx, f_attn, nullptr, 0, nullptr, f_dinv,
         n, DHD, MPADF, MPADF, MPADF, DM, 0, 0, (int)BH, NHEADS,
         (long long)NHEADS * n * MPADF, (long long)n * MPADF,
         (long long)NHEADS * DHD * MPADF, (long long)DHD * MPADF,
         (long long)n * DM, DHD, 0, 0, (long long)NHEADS * n, (long long)n);

    // ---- x = x + attn @ wo + bo
    conv(f_attn, bf_opA, BN * DM, 1.f);
    gemm(bf_opA, bf_wo, f_x2, bo + i * DM, 1, f_x1, nullptr,
         (int)BN, DM, DM, DM, DM, DM, 0, 0, 1, 1, 0,0,0,0,0,0,0,0,0,0);

    // ---- FFN: x = x + gelu(LN2(x) @ w1 + b1) @ w2 + b2
    k_layernorm<<<(int)BN, 256, 0, stream>>>(f_x2, bf_h, ln2s + i * DM, ln2b + i * DM);
    transp(ffw1 + (long long)i * DM * FFD, bf_f1, DM, FFD, FFD, DM, 1.f, 1, 1, 0, 0, 0);
    transp(ffw2 + (long long)i * FFD * DM, bf_f2, FFD, DM, DM, FFD, 1.f, 1, 1, 0, 0, 0);
    gemm(bf_h, bf_f1, bf_ff, ffb1 + i * FFD, 1, nullptr, nullptr,
         (int)BN, FFD, DM, DM, DM, FFD, 1, 1, 1, 1, 0,0,0,0,0,0,0,0,0,0);
    gemm(bf_ff, bf_f2, f_x1, ffb2 + i * DM, 1, f_x2, nullptr,
         (int)BN, DM, FFD, FFD, FFD, DM, 0, 0, 1, 1, 0,0,0,0,0,0,0,0,0,0);

    // ---- sequence squeeze: x[b,m,d] = sum_n lin_w[n,m] x[b,n,d] + lin_b[m]
    const float* lw = (const float*)d_in[20 + 2 * i];
    const float* lb = (const float*)d_in[21 + 2 * i];
    transp(lw, bf_lin, n, nout, nout, npad, 1.f, 1, 1, 0, 0, 0);
    transp(f_x1, bf_xT, n, DM, DM, npad, 1.f, BD, 1,
           (long long)n * DM, 0, (long long)DM * npad);
    const bool last = (i == DEPTH - 1);
    gemm(bf_lin, bf_xT, last ? d_out : (void*)f_x1, lb, 2, nullptr, nullptr,
         nout, DM, npad, npad, npad, DM, last ? 2 : 1, 0, BD, 1,
         0, 0, (long long)DM * npad, 0, (long long)nout * DM, 0, 0, 0, 0, 0);
  }
}
